// GCNLinkAutoencoder_47433618817228
// MI455X (gfx1250) — matrix operand order repaired and checked by reading, not compiled, not run on hardware
//
#include <hip/hip_runtime.h>
#include <hip/hip_bf16.h>

typedef __attribute__((ext_vector_type(16))) __bf16 v16bf;
typedef __attribute__((ext_vector_type(8)))  float  v8f;

#define NEG_SLOPE 0.01f

// ---------------------------------------------------------------- utilities
__global__ __launch_bounds__(256) void k_fill(float* __restrict__ p, float v, int n) {
    int t = blockIdx.x * blockDim.x + threadIdx.x;
    if (t < n) p[t] = v;
}

__global__ __launch_bounds__(256) void k_deg_scatter(const int* __restrict__ dst,
                                                     float* __restrict__ deg, int E) {
    int t = blockIdx.x * blockDim.x + threadIdx.x;
    if (t < E) atomicAdd(&deg[dst[t]], 1.0f);
}

__global__ __launch_bounds__(256) void k_rsqrt_inplace(float* __restrict__ d, int n) {
    int t = blockIdx.x * blockDim.x + threadIdx.x;
    if (t < n) d[t] = rsqrtf(d[t]);
}

// ------------------------------------------------- weight pre-swizzle (bf16)
// Wsw[(ct*KTILES + kt)*32 + lane][j] = bf16( W[kt*32 + (lane>>4)*16 + j][ct*16 + (lane&15)] )
// so each lane's B fragment for (ct,kt) is one contiguous, 32B-aligned v16bf.
__global__ __launch_bounds__(256) void k_swizzle_w(const float* __restrict__ W,
                                                   __bf16* __restrict__ Wsw,
                                                   int KTILES, int Nout, int total) {
    int t = blockIdx.x * blockDim.x + threadIdx.x;
    if (t >= total) return;
    int j    = t & 15;
    int lane = (t >> 4) & 31;
    int tile = t >> 9;               // ct*KTILES + kt
    int kt   = tile % KTILES;
    int ct   = tile / KTILES;
    int r = lane & 15, h = lane >> 4;
    int krow = kt * 32 + h * 16 + j;
    int col  = ct * 16 + r;
    Wsw[t] = (__bf16)W[(size_t)krow * Nout + col];
}

// ---------------------------------------------------------------- WMMA GEMM
// Out[M,Nout] = X[M,K] @ W[K,Nout]  (+ bias[Nout]) (+ resid[M,Nout])
// One wave per 16x16 output tile; K = KTILES*32 fully unrolled.
// A fragment built from fp32 X on the fly; B fragment is a single pre-swizzled
// 32-byte bf16 load per k-tile.
template <int KTILES, bool HAS_BIAS, bool HAS_RESID>
__global__ __launch_bounds__(32) void k_gemm_wmma_bf16(
    const float* __restrict__ X, const __bf16* __restrict__ Wsw,
    float* __restrict__ Out, int Nout,
    const float* __restrict__ bias, const float* __restrict__ resid) {

    const int lane = threadIdx.x;
    const int r    = lane & 15;
    const int h    = lane >> 4;
    const int K    = KTILES * 32;
    const int n    = blockIdx.y * 16 + r;

    const float* xrow   = X + (size_t)(blockIdx.x * 16 + r) * K + h * 16;
    const v16bf* wfrag  = (const v16bf*)Wsw + (size_t)blockIdx.y * (KTILES * 32) + lane;

    v8f acc = {};
#pragma unroll
    for (int kt = 0; kt < KTILES; ++kt) {
        v16bf a;
#pragma unroll
        for (int j = 0; j < 8; ++j) a[j]     = (__bf16)xrow[j];
#pragma unroll
        for (int j = 0; j < 8; ++j) a[8 + j] = (__bf16)xrow[8 + j];
        v16bf b = wfrag[(size_t)kt * 32];

        acc = __builtin_amdgcn_wmma_f32_16x16x32_bf16(
            false, a, false, b, (short)0, acc, false, false);

        xrow += 32;
    }

    const float bv = HAS_BIAS ? bias[n] : 0.0f;
    size_t base = (size_t)(blockIdx.x * 16 + h * 8) * Nout + n;
#pragma unroll
    for (int i = 0; i < 8; ++i) {
        float v = acc[i] + bv;
        if constexpr (HAS_RESID) v += resid[base];
        Out[base] = v;
        base += (size_t)Nout;
    }
}

// ---------------------------------------------------------------- GCN pieces
// agg[i,:] = h[i,:] * dinv[i]^2 + bias[:]   (self-loop + bias), float4-wide
__global__ __launch_bounds__(256) void k_init_self(
    const float4* __restrict__ h, const float* __restrict__ dinv,
    const float4* __restrict__ bias, float4* __restrict__ agg,
    int total4, int shift4, int mask4) {
    int t = blockIdx.x * blockDim.x + threadIdx.x;
    if (t < total4) {
        float di = dinv[t >> shift4];
        float s  = di * di;
        float4 hv = h[t];
        float4 bv = bias[t & mask4];
        float4 o;
        o.x = hv.x * s + bv.x;  o.y = hv.y * s + bv.y;
        o.z = hv.z * s + bv.z;  o.w = hv.w * s + bv.w;
        agg[t] = o;
    }
}

// agg[dst[e],f] += h[src[e],f] * dinv[src]*dinv[dst]
__global__ __launch_bounds__(128) void k_scatter_edges(
    const float* __restrict__ h, const float* __restrict__ dinv,
    const int* __restrict__ src, const int* __restrict__ dst,
    float* __restrict__ agg, int E, int F) {
    int e0 = blockIdx.x * 4;
#pragma unroll
    for (int i = 0; i < 4; ++i) {
        int e = e0 + i;
        if (e >= E) break;
        int s = src[e], d = dst[e];
        if (e + 1 < E) {   // hide next gather behind this edge's atomics
            const float* nxt = h + (size_t)src[e + 1] * F;
            __builtin_prefetch(nxt + threadIdx.x, 0, 0);
        }
        float nrm = dinv[s] * dinv[d];
        const float* hs = h + (size_t)s * F;
        float* ad = agg + (size_t)d * F;
        for (int f = threadIdx.x; f < F; f += 128)
            atomicAdd(&ad[f], hs[f] * nrm);
    }
}

// out = leaky(in) (+ resid), float4-wide
template <bool HAS_RESID>
__global__ __launch_bounds__(256) void k_leaky(
    const float4* __restrict__ in, const float4* __restrict__ resid,
    float4* __restrict__ out, int total4) {
    int t = blockIdx.x * blockDim.x + threadIdx.x;
    if (t < total4) {
        float4 v = in[t];
        v.x = (v.x >= 0.0f) ? v.x : NEG_SLOPE * v.x;
        v.y = (v.y >= 0.0f) ? v.y : NEG_SLOPE * v.y;
        v.z = (v.z >= 0.0f) ? v.z : NEG_SLOPE * v.z;
        v.w = (v.w >= 0.0f) ? v.w : NEG_SLOPE * v.w;
        if constexpr (HAS_RESID) {
            float4 rv = resid[t];
            v.x += rv.x; v.y += rv.y; v.z += rv.z; v.w += rv.w;
        }
        out[t] = v;
    }
}

// ---------------------------------------------------------------- driver
extern "C" void kernel_launch(void* const* d_in, const int* in_sizes, int n_in,
                              void* d_out, int out_size, void* d_ws, size_t ws_size,
                              hipStream_t stream) {
    const float* x  = (const float*)d_in[0];
    const int*   ei = (const int*)d_in[1];
    const float* W1 = (const float*)d_in[2];  const float* b1 = (const float*)d_in[3];
    const float* W2 = (const float*)d_in[4];  const float* b2 = (const float*)d_in[5];
    const float* W3 = (const float*)d_in[6];  const float* b3 = (const float*)d_in[7];
    const float* W4 = (const float*)d_in[8];  const float* b4 = (const float*)d_in[9];
    const float* W5 = (const float*)d_in[10]; const float* b5 = (const float*)d_in[11];
    const float* Wd = (const float*)d_in[12]; const float* bd = (const float*)d_in[13];

    const int IN  = 128, HID = 128, LAT = 64;
    const int N = in_sizes[0] / IN;          // 50000
    const int E = in_sizes[1] / 2;           // 800000
    const int* srcE = ei;
    const int* dstE = ei + E;

    float* ws   = (float*)d_ws;
    const size_t NF = (size_t)N * HID;       // 6.4M floats
    float* dinv = ws;                        // [N]
    float* h    = ws + 51200;                // [N,128] gemm result
    float* agg  = h + NF;                    // [N,128] aggregation buffer
    float* bufA = agg + NF;                  // x1 / x3
    float* bufB = bufA + NF;                 // x2 / xr
    // pre-swizzled bf16 weights (each slot 8192 floats = 16384 bf16)
    __bf16* W1s = (__bf16*)(bufB + NF);
    __bf16* W2s = W1s + 16384;
    __bf16* W3s = W2s + 16384;
    __bf16* W4s = W3s + 16384;               // uses 8192
    __bf16* W5s = W4s + 16384;               // uses 8192
    __bf16* Wds = W5s + 16384;

    float* rec = (float*)d_out;              // [N,128]
    float* z   = rec + NF;                   // [N,64]

    const int TB = 256;
    // weight pre-swizzle (tiny; L2 resident afterwards)
    k_swizzle_w<<<(128 * 128 + TB - 1) / TB, TB, 0, stream>>>(W1, W1s, 4, 128, 128 * 128);
    k_swizzle_w<<<(128 * 128 + TB - 1) / TB, TB, 0, stream>>>(W2, W2s, 4, 128, 128 * 128);
    k_swizzle_w<<<(128 * 128 + TB - 1) / TB, TB, 0, stream>>>(W3, W3s, 4, 128, 128 * 128);
    k_swizzle_w<<<(128 * 64  + TB - 1) / TB, TB, 0, stream>>>(W4, W4s, 4, 64,  128 * 64);
    k_swizzle_w<<<(64  * 128 + TB - 1) / TB, TB, 0, stream>>>(W5, W5s, 2, 128, 64 * 128);
    k_swizzle_w<<<(128 * 128 + TB - 1) / TB, TB, 0, stream>>>(Wd, Wds, 4, 128, 128 * 128);

    // degrees: deg[i] = 1 (self loop) + #(dst==i); dinv = rsqrt(deg)
    k_fill<<<(N + TB - 1) / TB, TB, 0, stream>>>(dinv, 1.0f, N);
    k_deg_scatter<<<(E + TB - 1) / TB, TB, 0, stream>>>(dstE, dinv, E);
    k_rsqrt_inplace<<<(N + TB - 1) / TB, TB, 0, stream>>>(dinv, N);

    auto aggregate = [&](const float* bm, int Fout, float* aggDst) {
        int total4 = N * Fout / 4;
        int shift4 = (Fout == 128) ? 5 : 4;          // vec4 per row: 32 or 16
        k_init_self<<<(total4 + TB - 1) / TB, TB, 0, stream>>>(
            (const float4*)h, dinv, (const float4*)bm, (float4*)aggDst,
            total4, shift4, (Fout / 4) - 1);
        k_scatter_edges<<<(E + 3) / 4, 128, 0, stream>>>(h, dinv, srcE, dstE,
                                                         aggDst, E, Fout);
    };
    auto conv128 = [&](const float* Xin, const __bf16* Wm, const float* bm,
                       int Fout, float* aggDst) {
        dim3 g(N / 16, Fout / 16);
        k_gemm_wmma_bf16<4, false, false><<<g, 32, 0, stream>>>(Xin, Wm, h, Fout,
                                                                nullptr, nullptr);
        aggregate(bm, Fout, aggDst);
    };

    const int tot4 = (int)(NF / 4);
    // layer 1: x1 = leaky(conv(x))
    conv128(x, W1s, b1, HID, agg);
    k_leaky<false><<<(tot4 + TB - 1) / TB, TB, 0, stream>>>(
        (const float4*)agg, nullptr, (float4*)bufA, tot4);
    // layer 2: x2 = leaky(conv(x1))
    conv128(bufA, W2s, b2, HID, agg);
    k_leaky<false><<<(tot4 + TB - 1) / TB, TB, 0, stream>>>(
        (const float4*)agg, nullptr, (float4*)bufB, tot4);
    // layer 3: x3 = leaky(conv(x2)) + x2
    conv128(bufB, W3s, b3, HID, agg);
    k_leaky<true><<<(tot4 + TB - 1) / TB, TB, 0, stream>>>(
        (const float4*)agg, (const float4*)bufB, (float4*)bufA, tot4);
    // layer 4: z = conv(x3)   (no activation; aggregate straight into d_out)
    conv128(bufA, W4s, b4, LAT, z);
    // layer 5: xr = leaky(conv(z)), K = 64
    {
        dim3 g(N / 16, HID / 16);
        k_gemm_wmma_bf16<2, false, false><<<g, 32, 0, stream>>>(z, W5s, h, HID,
                                                                nullptr, nullptr);
        aggregate(b5, HID, agg);
        k_leaky<false><<<(tot4 + TB - 1) / TB, TB, 0, stream>>>(
            (const float4*)agg, nullptr, (float4*)bufB, tot4);
    }
    // decode: rec = xr @ Wd + bd + x   (bias + residual fused into WMMA epilogue)
    dim3 gd(N / 16, IN / 16);
    k_gemm_wmma_bf16<4, true, true><<<gd, 32, 0, stream>>>(bufB, Wds, rec, IN, bd, x);
}
